// Modulated_conv2d_layer_10746008174668
// MI455X (gfx1250) — matrix operand order repaired and checked by reading, not compiled, not run on hardware
//
#include <hip/hip_runtime.h>
#include <hip/hip_bf16.h>

typedef __attribute__((ext_vector_type(16))) _Float16 v16h;
typedef __attribute__((ext_vector_type(8)))  float    v8f;
typedef _Float16 half_t;

#define B_   8
#define IC   512
#define OC   512
#define DD   512
#define HH   64
#define WW   64

#define DENSE_SCALE 0.04419417382415922f      // 1/sqrt(512)
#define CONV_SCALE  0.014731391274719738f     // 1/sqrt(512*9)

// workspace layout (bytes, all 16B aligned)
#define WS_STYLE 0u
#define WS_D     16384u
#define WS_W2    32768u
#define WS_WP    (32768u + 1048576u)                 // 1,081,344
#define WS_XS    (WS_WP + 9u*512u*512u*2u)           // 5,799,936  (+33.5MB)

// ---------------- style: [B,I] = dlatents @ dense_w^T * s + b ----------------
__global__ __launch_bounds__(256) void style_kernel(const float* __restrict__ dl,
                                                    const float* __restrict__ dw,
                                                    const float* __restrict__ db,
                                                    float* __restrict__ style) {
    int t = blockIdx.x * 256 + threadIdx.x;          // 0..4095
    if (t >= B_ * IC) return;
    int b = t >> 9, i = t & 511;
    const float* dlrow = dl + (size_t)b * DD;
    const float* dwrow = dw + (size_t)i * DD;
    float s = 0.f;
    #pragma unroll 4
    for (int k = 0; k < DD; ++k) s += dlrow[k] * dwrow[k];
    style[t] = s * DENSE_SCALE + db[i];
}

// -------- wprep: w_p[tap][o][i] = f16(w[o][i][tap]*cs);  w2[o][i] = sum_k w^2 --------
__global__ __launch_bounds__(256) void wprep_kernel(const float* __restrict__ w,
                                                    float* __restrict__ w2,
                                                    half_t* __restrict__ wp) {
    int t = blockIdx.x * 256 + threadIdx.x;          // 0..262143  (= o*512+i)
    if (t >= OC * IC) return;
    const float* wsrc = w + (size_t)t * 9;           // [O][I][3][3] layout
    float s2 = 0.f;
    #pragma unroll
    for (int k = 0; k < 9; ++k) {
        float v = wsrc[k];
        s2 += v * v;
        wp[(size_t)k * OC * IC + t] = (half_t)(v * CONV_SCALE);
    }
    w2[t] = s2;
}

// -------- demod: d[b][o] = rsqrt(sum_i w2[o][i]*style^2 * cs^2 + 1e-8) --------
__global__ __launch_bounds__(256) void demod_kernel(const float* __restrict__ w2,
                                                    const float* __restrict__ style,
                                                    float* __restrict__ dmod) {
    int t = blockIdx.x * 256 + threadIdx.x;          // 0..4095
    if (t >= B_ * OC) return;
    int b = t >> 9, o = t & 511;
    const float* w2row = w2 + (size_t)o * IC;
    const float* srow  = style + (size_t)b * IC;
    float acc = 0.f;
    #pragma unroll 4
    for (int i = 0; i < IC; ++i) { float s = srow[i]; acc += w2row[i] * s * s; }
    dmod[t] = rsqrtf(acc * (CONV_SCALE * CONV_SCALE) + 1e-8f);
}

// -------- xs: f16(x * style), vectorized --------
__global__ __launch_bounds__(256) void xs_kernel(const float* __restrict__ x,
                                                 const float* __restrict__ style,
                                                 half_t* __restrict__ xs) {
    size_t idx = (size_t)blockIdx.x * 256 + threadIdx.x;  // one float4 each
    size_t e0 = idx * 4;
    if (e0 >= (size_t)B_ * IC * HH * WW) return;
    unsigned bi = (unsigned)(e0 >> 12);              // /(64*64) -> b*512+i
    float s = style[bi];
    float4 v = *(const float4*)(x + e0);
    union { half_t h[4]; uint2 u; } p;
    p.h[0] = (half_t)(v.x * s); p.h[1] = (half_t)(v.y * s);
    p.h[2] = (half_t)(v.z * s); p.h[3] = (half_t)(v.w * s);
    *(uint2*)(xs + e0) = p.u;
}

// ---------------- main conv: WMMA implicit GEMM ----------------
// grid.x = b*64+h (512), grid.y = o-block (4 x 128). 256 threads = 8 wave32.
// Wave wv owns output rows o0+wv*16..+15, all 64 cols (4 N-tiles of 16).
#define KB      32
#define LDS_POS 66
#define LDS_CH  40            // padded (only 32 used): 80B stride -> conflict-free b128
#define NPAIR   (3 * 32 * 32) // 3072 u32 staging slots (2 halfs each)
#define NITER   12            // 3072 / 256

struct Frag { uint4 q0, q1; };

__global__ __launch_bounds__(256) void conv_kernel(const half_t* __restrict__ xs,
                                                   const half_t* __restrict__ wp,
                                                   const float* __restrict__ dmod,
                                                   float* __restrict__ out) {
    __shared__ __align__(16) half_t lds[2][3][LDS_POS][LDS_CH];

    const int tid   = threadIdx.x;
    const int bh    = blockIdx.x;
    const int b     = bh >> 6;
    const int h     = bh & 63;
    const int o0    = blockIdx.y * 128;
    const int wv    = tid >> 5;
    const int lane  = tid & 31;
    const int col   = lane & 15;          // M for A-lanes, N for B/C-lanes
    const int lhalf = lane >> 4;          // 0/1

    // halo columns (pos 0 and 65) are zero for every channel/k-step: zero once
    #pragma unroll
    for (int e = tid; e < 384; e += 256) {
        int ch = e & 31;
        int r  = e >> 5;                  // 0..11
        int ps = (r & 1) ? (LDS_POS - 1) : 0;
        int dh = (r >> 1) % 3;
        int bufi = r / 6;
        lds[bufi][dh][ps][ch] = (half_t)0.f;
    }

    const v8f vzero = {0.f, 0.f, 0.f, 0.f, 0.f, 0.f, 0.f, 0.f};
    v8f acc[4];
    #pragma unroll
    for (int n = 0; n < 4; ++n) acc[n] = vzero;

    unsigned sval[NITER];

    // power-of-2 staging: interior positions only (wx 0..63), u32 = 2 halfs
    auto stage_load = [&](int i0) {
        #pragma unroll
        for (int it = 0; it < NITER; ++it) {
            int slot = tid + it * 256;
            int wpair = slot & 31;                 // wx = 2*wpair
            int ch    = (slot >> 5) & 31;
            int dh    = slot >> 10;                // 0..2
            int hy    = h + dh - 1;
            unsigned v = 0u;
            if ((unsigned)hy < 64u)
                v = *(const unsigned*)(xs + (((size_t)b * IC + (i0 + ch)) * HH + hy) * WW
                                           + 2 * wpair);
            sval[it] = v;
        }
    };
    auto stage_store = [&](int buf) {
        #pragma unroll
        for (int it = 0; it < NITER; ++it) {
            int slot = tid + it * 256;
            int wpair = slot & 31;
            int ch    = (slot >> 5) & 31;
            int dh    = slot >> 10;
            union { unsigned u; half_t hh[2]; } c;
            c.u = sval[it];
            lds[buf][dh][1 + 2 * wpair][ch] = c.hh[0];
            lds[buf][dh][2 + 2 * wpair][ch] = c.hh[1];
        }
    };

    // fragment loaders
    auto loadA = [&](int tap, int i0) -> Frag {
        // A 16x32 f16: lane holds row M=col; K halfs [kb..kb+7],[kb+16..kb+23], kb=lhalf*8
        const half_t* ap = wp + (((size_t)tap * OC) + o0 + wv * 16 + col) * IC
                              + i0 + lhalf * 8;
        Frag f;
        f.q0 = *(const uint4*)(ap);
        f.q1 = *(const uint4*)(ap + 16);
        return f;
    };
    auto loadB = [&](int buf, int dh, int dw, int n) -> Frag {
        // B 32x16 f16: lane holds col N=col; K channels [kb..kb+7],[kb+16..kb+23], kb=lhalf*8
        const half_t* bp = &lds[buf][dh][16 * n + col + dw][lhalf * 8];
        Frag f;
        f.q0 = *(const uint4*)(bp);
        f.q1 = *(const uint4*)(bp + 16);
        return f;
    };
    auto mma = [&](const Frag& a, const Frag& bb, v8f c) -> v8f {
        union { Frag f; v16h v; } ua, ub;
        ua.f = a; ub.f = bb;
        return __builtin_amdgcn_wmma_f32_16x16x32_f16(false, ua.v, false, ub.v,
                                                      (short)0, c, false, false);
    };

    stage_load(0);
    stage_store(0);

    for (int ks = 0; ks < IC / KB; ++ks) {
        const int buf = ks & 1;
        const int i0  = ks * KB;
        __syncthreads();
        if (ks + 1 < IC / KB) stage_load((ks + 1) * KB);   // prefetch into regs

        // software-pipelined 9-tap loop: load tap t+1 fragments while tap t's
        // 4 WMMAs issue back-to-back (distinct B register sets -> no per-WMMA wait)
        Frag Acur = loadA(0, i0);
        Frag Bcur[4];
        #pragma unroll
        for (int n = 0; n < 4; ++n) Bcur[n] = loadB(buf, 0, 0, n);

        #pragma unroll
        for (int tap = 0; tap < 9; ++tap) {
            Frag Anxt;
            Frag Bnxt[4];
            if (tap < 8) {
                const int dh = (tap + 1) / 3, dw = (tap + 1) % 3;
                Anxt = loadA(tap + 1, i0);
                #pragma unroll
                for (int n = 0; n < 4; ++n) Bnxt[n] = loadB(buf, dh, dw, n);
            }
            #pragma unroll
            for (int n = 0; n < 4; ++n) acc[n] = mma(Acur, Bcur[n], acc[n]);
            if (tap < 8) {
                Acur = Anxt;
                #pragma unroll
                for (int n = 0; n < 4; ++n) Bcur[n] = Bnxt[n];
            }
        }
        if (ks + 1 < IC / KB) stage_store((ks + 1) & 1);
    }

    // epilogue: scale by demod and store (C layout: VGPR r -> M=r+lhalf*8, lane -> N)
    const int moff = lhalf * 8;
    float dv[8];
    #pragma unroll
    for (int r = 0; r < 8; ++r)
        dv[r] = dmod[b * OC + o0 + wv * 16 + moff + r];
    #pragma unroll
    for (int n = 0; n < 4; ++n) {
        #pragma unroll
        for (int r = 0; r < 8; ++r) {
            int o = o0 + wv * 16 + moff + r;
            out[(((size_t)b * OC + o) * HH + h) * WW + 16 * n + col] = acc[n][r] * dv[r];
        }
    }
}

// ---------------- launch ----------------
extern "C" void kernel_launch(void* const* d_in, const int* in_sizes, int n_in,
                              void* d_out, int out_size, void* d_ws, size_t ws_size,
                              hipStream_t stream) {
    const float* x  = (const float*)d_in[0];   // [8,512,64,64]
    const float* dl = (const float*)d_in[1];   // [8,512]
    const float* w  = (const float*)d_in[2];   // [512,512,3,3]
    const float* dw = (const float*)d_in[3];   // [512,512]
    const float* db = (const float*)d_in[4];   // [512]
    float* out = (float*)d_out;

    char* ws = (char*)d_ws;
    float*  style = (float*)(ws + WS_STYLE);
    float*  dmod  = (float*)(ws + WS_D);
    float*  w2    = (float*)(ws + WS_W2);
    half_t* wp    = (half_t*)(ws + WS_WP);
    half_t* xs    = (half_t*)(ws + WS_XS);

    style_kernel<<<16, 256, 0, stream>>>(dl, dw, db, style);
    wprep_kernel<<<1024, 256, 0, stream>>>(w, w2, wp);
    demod_kernel<<<16, 256, 0, stream>>>(w2, style, dmod);
    xs_kernel<<<16384, 256, 0, stream>>>(x, style, xs);

    dim3 grid(512, 4);
    conv_kernel<<<grid, 256, 0, stream>>>(xs, wp, dmod, out);
}
